// MetaUpscaleModule_42468636623611
// MI455X (gfx1250) — matrix operand order repaired and checked by reading, compiled, not run on hardware
//
#include <hip/hip_runtime.h>
#include <hip/hip_bf16.h>

// MetaUpscale fused kernels for MI455X (gfx1250), wave32 + WMMA bf16.
//   K1: hmid = relu(v@w1+b1)       -> bf16, A-fragment-packed          (ws)
//   K2: w2   -> bf16, B-fragment-packed                                (ws)
//   K3: unfold(F_LR) patches       -> bf16, A-fragment-packed compact  (ws)
//   K4: per 32-row tile: Wtile = hmid@w2 + b2 (WMMA stage 1, LDS-staged),
//       out += pat (x) Wtile (WMMA stage 2), write fp32 output.

typedef __attribute__((ext_vector_type(16))) __bf16 v16bf;
typedef __attribute__((ext_vector_type(8)))  float  v8f;

#define NB    8      // batch
#define CIN   32     // input channels
#define HH    48
#define WW    48
#define HID   256
#define FEAT  288    // C*K*K
#define OCH   32
#define NW    9216   // FEAT*OCH == W_pred columns
#define NPIX  9216   // H*S*W*S   == W_pred rows

#define PAT_ELEMS   (2304*9*256)        // pixels * fchunks * (16 lanes * 16 e)
#define PAT_ZERO    (2304*9*16)         // v16bf index of the shared zero frag

__device__ __forceinline__ unsigned short f2bf(float x) {   // RNE (pack kernels)
  union { float f; unsigned u; } c; c.f = x;
  unsigned u = c.u;
  return (unsigned short)((u + 0x7FFFu + ((u >> 16) & 1u)) >> 16);
}

__device__ __forceinline__ void store_bf16_fast(unsigned short* p, float x) {
  union { float f; unsigned u; } c; c.f = x;          // round-half-up: 1 VALU op,
  *p = (unsigned short)((c.u + 0x8000u) >> 16);       // shift folds into b16 store
}

// ---------------- K1: MLP layer 1, packed into 16-bit A-fragment layout -----
// A 16x32 (ISA 7.12.2): lane<16: M=lane, K = e<8 ? e : e+8
//                       lane>=16: M=lane-16, K pattern +8.
__global__ __launch_bounds__(256) void hmid_pack_kernel(
    const float* __restrict__ v, const float* __restrict__ w1,
    const float* __restrict__ b1, unsigned short* __restrict__ hmidPack)
{
  int idx = blockIdx.x * 256 + threadIdx.x;     // NPIX*HID threads
  int k = idx & (HID - 1);
  int r = idx >> 8;
  float a = b1[k]
          + v[r*3 + 0] * w1[0*HID + k]
          + v[r*3 + 1] * w1[1*HID + k]
          + v[r*3 + 2] * w1[2*HID + k];
  a = fmaxf(a, 0.0f);
  int rt16 = r >> 4, M = r & 15;
  int ks = k >> 5,  kc = k & 31;
  int lane_, e_;
  if      (kc <  8) { lane_ = M;      e_ = kc;      }
  else if (kc < 16) { lane_ = M + 16; e_ = kc - 8;  }
  else if (kc < 24) { lane_ = M;      e_ = kc - 8;  }
  else              { lane_ = M + 16; e_ = kc - 16; }
  hmidPack[((rt16*8 + ks)*32 + lane_)*16 + e_] = f2bf(a);
}

// ---------------- K2: w2 -> bf16, packed into B-fragment layout -------------
// B 32x16: lanes 0-15: N=lane, K = e<8 ? e : e+8; lanes 16-31: N=lane-16, K pattern +8.
__global__ __launch_bounds__(256) void w2_pack_kernel(
    const float* __restrict__ w2, unsigned short* __restrict__ w2Pack)
{
  int idx = blockIdx.x * 256 + threadIdx.x;     // HID*NW threads
  int col = idx % NW;
  int k   = idx / NW;
  int gtile = col >> 4, nl = col & 15;
  int ks = k >> 5,  kc = k & 31;
  int lane_ = ((kc >> 3) & 1) ? (nl + 16) : nl;
  int e_    = (kc & 7) + ((kc >= 16) ? 8 : 0);
  w2Pack[((gtile*8 + ks)*32 + lane_)*16 + e_] = f2bf(w2[idx]);
}

// ---------------- K3: unfold(F_LR) -> bf16 A-fragments (compact) ------------
// Per (pixel l, f-chunk fc): 16 "real" lanes (lane16 = hi*8 + n) x 16 elems.
// Lanes for batch rows 8..15 are served by one shared zero fragment.
__global__ __launch_bounds__(256) void pat_pack_kernel(
    const float* __restrict__ F_LR, unsigned short* __restrict__ patPack)
{
  int idx = blockIdx.x * 256 + threadIdx.x;
  if (idx >= PAT_ELEMS) {
    if (idx < PAT_ELEMS + 16) patPack[idx] = 0;   // zero fragment
    return;
  }
  int e      = idx & 15;
  int lane16 = (idx >> 4) & 15;
  int lf     = idx >> 8;            // l*9 + fc
  int fc     = lf % 9;
  int l      = lf / 9;
  int n  = lane16 & 7;              // batch row
  int hi = lane16 >> 3;             // wave half
  int fl = e + hi*16;                       // K within chunk, contiguous K order
  int f  = fc*32 + fl;              // 0..287
  int c  = f / 9;
  int r9 = f - c*9;
  int ki = r9 / 3;
  int kj = r9 - ki*3;
  int hh = l / WW, ww = l - hh*WW;
  int hs = hh + ki - 1, ws = ww + kj - 1;
  float val = 0.0f;
  if (hs >= 0 && hs < HH && ws >= 0 && ws < WW)
    val = F_LR[((n*CIN + c)*HH + hs)*WW + ws];
  patPack[idx] = f2bf(val);
}

// ---------------- K4: fused weight-prediction + per-pixel conv --------------
// One workgroup = 32 consecutive W_pred rows (fixed h, sh; 16 pixels x 2 sw).
// LDS: wchunk [32 r][32 o][32 f_local] bf16 = 64 KB (of 320 KB/WGP).
__global__ __launch_bounds__(256) void fused_upscale_kernel(
    const float* __restrict__ b2,
    const unsigned short* __restrict__ hmidPack,
    const unsigned short* __restrict__ w2Pack,
    const unsigned short* __restrict__ patPack,
    float* __restrict__ out)
{
  extern __shared__ unsigned short wchunk[];   // 32768 u16

  const int rt     = blockIdx.x;               // 288 tiles of 32 rows
  const int tid    = threadIdx.x;
  const int wave   = tid >> 5;
  const int lane   = tid & 31;
  const int lane15 = lane & 15;
  const bool hiHalf = lane >= 16;

  const int h     = rt / 6;
  const int rem   = (rt - h*6) * 32;           // 0..160
  const int sh    = rem / 96;
  const int wbase = (rem - sh*96) >> 1;        // 0,16,32

  const v16bf* hfrag_g = (const v16bf*)hmidPack;
  const v16bf* bfrag_g = (const v16bf*)w2Pack;
  const v16bf* pfrag_g = (const v16bf*)patPack;

  // hmid A-fragments for both 16-row M-tiles, K = 256 (8 k-steps): 128 VGPRs
  v16bf afrag[2][8];
#pragma unroll
  for (int m = 0; m < 2; ++m)
#pragma unroll
    for (int ks = 0; ks < 8; ++ks)
      afrag[m][ks] = hfrag_g[((rt*2 + m)*8 + ks)*32 + lane];

  // epilogue accumulators: 4 rows x 2 o-tiles per wave
  v8f eacc[4][2];
#pragma unroll
  for (int i = 0; i < 4; ++i)
#pragma unroll
    for (int j = 0; j < 2; ++j)
      eacc[i][j] = (v8f){0.f,0.f,0.f,0.f,0.f,0.f,0.f,0.f};

  for (int fc = 0; fc < 9; ++fc) {             // 9 chunks x 32 features
    // ---- issue epilogue A-fragment loads early (L2-resident, hides under
    //      stage-1 WMMAs); pad lanes (batch rows 8..15) read the zero frag ----
    v16bf pa[2];
#pragma unroll
    for (int pp = 0; pp < 2; ++pp) {
      int p16 = wave*2 + pp;
      int l   = h*WW + wbase + p16;
      int fi  = (l*9 + fc)*16 + (hiHalf ? 8 : 0) + lane15;
      pa[pp]  = pfrag_g[(lane15 < 8) ? fi : PAT_ZERO];
    }

    // prefetch next chunk's B fragments into cache (global_prefetch_b8)
    if (fc < 8)
      __builtin_prefetch((const void*)&bfrag_g[(((fc+1)*64 + wave*8)*8)*32 + lane], 0, 1);

    // ---- stage 1: Wtile[32r x 1024c] = hmid@w2 + b2, columns split 8 ways --
#pragma unroll 2
    for (int nt = 0; nt < 8; ++nt) {
      int gtile = fc*64 + wave*8 + nt;         // global 16-col tile
      float bb = b2[gtile*16 + lane15];        // bias preloaded into C operand
      v8f acc0 = (v8f){bb,bb,bb,bb,bb,bb,bb,bb};
      v8f acc1 = acc0;
#pragma unroll
      for (int ks = 0; ks < 8; ++ks) {
        v16bf bf = bfrag_g[(gtile*8 + ks)*32 + lane];
        acc0 = __builtin_amdgcn_wmma_f32_16x16x32_bf16(
                   false, afrag[0][ks], false, bf, (short)0, acc0, false, false);
        acc1 = __builtin_amdgcn_wmma_f32_16x16x32_bf16(
                   false, afrag[1][ks], false, bf, (short)0, acc1, false, false);
      }
      // store to LDS transposed: wchunk[row][o][f_local], bf16
      int cl   = (wave*8 + nt)*16 + lane15;    // chunk-local column 0..1023
      int base = (cl & 31)*32 + (cl >> 5);     // o*32 + f_local
      int moff = hiHalf ? 8 : 0;               // C/D layout M offset
#pragma unroll
      for (int j = 0; j < 8; ++j) {
        store_bf16_fast(&wchunk[(j + moff)*1024      + base], acc0[j]);
        store_bf16_fast(&wchunk[(j + moff + 16)*1024 + base], acc1[j]);
      }
    }
    __syncthreads();

    // ---- stage 2: out[n,o] += pat[n,f] * W[r,f,o] via WMMA (M=batch pad 16)
#pragma unroll
    for (int pp = 0; pp < 2; ++pp) {
#pragma unroll
      for (int ss = 0; ss < 2; ++ss) {         // sw = 0/1 share pixel patches
        int rloc = (wave*2 + pp)*2 + ss;       // row within 32-row tile
#pragma unroll
        for (int ot = 0; ot < 2; ++ot) {
          int o = ot*16 + lane15;
          v16bf pb = *(const v16bf*)(wchunk + rloc*1024 + o*32 + (hiHalf ? 16 : 0));
          eacc[pp*2 + ss][ot] = __builtin_amdgcn_wmma_f32_16x16x32_bf16(
              false, pa[pp], false, pb, (short)0, eacc[pp*2 + ss][ot], false, false);
        }
      }
    }
    __syncthreads();
  }

  // ---- write fp32 output: out[n, o, h*2+sh, w*2+sw] ----
  if (!hiHalf) {                               // lanes 0-15 hold batch rows 0..7
    int h2 = h*2 + sh;
#pragma unroll
    for (int rr = 0; rr < 4; ++rr) {
      int rloc = wave*4 + rr;
      int w2c  = (wbase + (rloc >> 1))*2 + (rloc & 1);
#pragma unroll
      for (int ot = 0; ot < 2; ++ot) {
        int o = ot*16 + lane15;
        v8f a = eacc[rr][ot];
#pragma unroll
        for (int j = 0; j < 8; ++j)            // j = batch n
          out[((j*OCH + o)*96 + h2)*96 + w2c] = a[j];
      }
    }
  }
}

// ---------------------------------------------------------------------------
extern "C" void kernel_launch(void* const* d_in, const int* in_sizes, int n_in,
                              void* d_out, int out_size, void* d_ws, size_t ws_size,
                              hipStream_t stream) {
  (void)in_sizes; (void)n_in; (void)out_size; (void)ws_size;
  const float* F_LR = (const float*)d_in[0];
  const float* v    = (const float*)d_in[1];
  const float* w1   = (const float*)d_in[2];
  const float* b1   = (const float*)d_in[3];
  const float* w2   = (const float*)d_in[4];
  const float* b2   = (const float*)d_in[5];
  float* out = (float*)d_out;

  const size_t HMID_BYTES = (size_t)NPIX * HID * 2;      // 4,718,592
  const size_t W2_BYTES   = (size_t)HID * NW * 2;        // 4,718,592
  unsigned short* hmidPack = (unsigned short*)d_ws;
  unsigned short* w2Pack   = (unsigned short*)((char*)d_ws + HMID_BYTES);
  unsigned short* patPack  = (unsigned short*)((char*)d_ws + HMID_BYTES + W2_BYTES);

  hipLaunchKernelGGL(hmid_pack_kernel, dim3(NPIX*HID/256), dim3(256), 0, stream,
                     v, w1, b1, hmidPack);
  hipLaunchKernelGGL(w2_pack_kernel, dim3(HID*NW/256), dim3(256), 0, stream,
                     w2, w2Pack);
  hipLaunchKernelGGL(pat_pack_kernel, dim3((PAT_ELEMS + 16 + 255)/256), dim3(256),
                     0, stream, F_LR, patPack);
  hipLaunchKernelGGL(fused_upscale_kernel, dim3(288), dim3(256),
                     /*dynamic LDS*/ 65536, stream,
                     b2, hmidPack, w2Pack, patPack, out);
}
